// GCN_18305150616172
// MI455X (gfx1250) — compile-verified, matrix operand order checked by reading, not run on hardware
//
#include <hip/hip_runtime.h>
#include <hip/hip_bf16.h>

#define N_NODES   100000
#define N_EDGES   3200000
#define N_GRAPHS  64
#define IN_FEATS  128
#define NUM_CLASSES 10

typedef __attribute__((ext_vector_type(16))) _Float16 v16h;
typedef __attribute__((ext_vector_type(8)))  float    v8f;

// ---------------------------------------------------------------------------
// Utility: zero fill (grid-stride)
// ---------------------------------------------------------------------------
__global__ void zero_f32(float* __restrict__ p, long long n) {
    for (long long i = blockIdx.x * (long long)blockDim.x + threadIdx.x;
         i < n; i += (long long)gridDim.x * blockDim.x)
        p[i] = 0.0f;
}

// ---------------------------------------------------------------------------
// Degree accumulation: deg_out[src[e]] += 1, deg_in[dst[e]] += 1
// ---------------------------------------------------------------------------
__global__ void degree_kernel(const int* __restrict__ src, const int* __restrict__ dst,
                              float* __restrict__ deg_out, float* __restrict__ deg_in,
                              int n_edges) {
    int e = blockIdx.x * blockDim.x + threadIdx.x;
    if (e >= n_edges) return;
    atomicAdd(&deg_out[src[e]], 1.0f);
    atomicAdd(&deg_in[dst[e]], 1.0f);
}

// norm = rsqrt(max(deg, 1))
__global__ void rsqrt_kernel(const float* __restrict__ deg, float* __restrict__ norm, int n) {
    int i = blockIdx.x * blockDim.x + threadIdx.x;
    if (i >= n) return;
    norm[i] = rsqrtf(fmaxf(deg[i], 1.0f));
}

// ---------------------------------------------------------------------------
// Weight pre-pack: W[K x N] f32 (row-major) -> f16 in the exact WMMA B-operand
// register image.  Layout: P[((c*NT + nt)*32 + lane)*16 + i], where c = K/32
// chunk, nt = 16-col tile, and per ISA 7.12.2 (16-bit B 32x16):
//   lane l supplies column N = nt*16 + (l&15);
//   lanes 0-15 hold K = {0..7, 16..23}, lanes 16-31 hold K = {8..15, 24..31}
//   within the chunk; halves i=0..7 -> first K-octet, i=8..15 -> second.
// Each GEMM lane then loads its 16 halves as one contiguous 32B vector.
// ---------------------------------------------------------------------------
template<int K, int N>
__global__ void pack_weights(const float* __restrict__ W, _Float16* __restrict__ P) {
    constexpr int NT = N / 16;
    int tid = blockIdx.x * blockDim.x + threadIdx.x;
    if (tid >= K * N) return;
    const int i    = tid & 15;
    const int lane = (tid >> 4) & 31;
    const int t    = tid >> 9;          // t = c*NT + nt
    const int nt   = t % NT;
    const int c    = t / NT;
    const int khalf = (lane >> 4) << 3; // 0 or 8
    const int kk  = c * 32 + ((i < 8) ? (khalf + i) : (16 + khalf + (i - 8)));
    const int col = nt * 16 + (lane & 15);
    P[tid] = (_Float16)W[kk * N + col];
}

// ---------------------------------------------------------------------------
// WMMA GEMM: C[M x (NT*16)] = epilogue( (scale_row ⊙ A[M x K]) @ W )
// One wave32 computes a full 16-row strip (NT accumulators): A is loaded and
// converted ONCE per K-chunk and reused across NT v_wmma issues.  K and NT are
// compile-time -> fully unrolled, (K/32)*NT static WMMAs.
// A layout (16-bit A 16x32, ISA 7.12.2): lane l holds row M=(l&15);
//   lanes 0-15 hold K = {0..7, 16..23}, lanes 16-31 hold K = {8..15, 24..31}.
// C/D layout: VGPR v, lane l -> row = v + 8*(l>>4), col = (l&15).
// ---------------------------------------------------------------------------
template<int K, int NT>
__global__ void gemm_wmma(const float* __restrict__ A, const _Float16* __restrict__ P,
                          const float* __restrict__ scale,   // per-row of A, may be null
                          const float* __restrict__ bias,    // per-col, may be null
                          float* __restrict__ C, int Mtiles, int relu) {
    constexpr int N = NT * 16;
    const int lane = threadIdx.x & 31;
    const int mt   = (blockIdx.x * blockDim.x + threadIdx.x) >> 5;
    if (mt >= Mtiles) return;               // wave-uniform: EXEC stays all-1s

    const int row   = (mt << 4) + (lane & 15);
    const int khalf = (lane >> 4) << 3;     // 0 or 8
    const float s   = scale ? scale[row] : 1.0f;
    const float* Arow = A + (size_t)row * K;
    const v16h*  Bp   = (const v16h*)P;     // 16-half (32B) granules

    v8f acc[NT];
#pragma unroll
    for (int nt = 0; nt < NT; ++nt) {
        v8f z = {};
        acc[nt] = z;
    }

#pragma unroll
    for (int c = 0; c < K / 32; ++c) {
        v16h a;
#pragma unroll
        for (int i = 0; i < 8; ++i) {
            a[i]     = (_Float16)(Arow[c * 32 + khalf + i] * s);
            a[8 + i] = (_Float16)(Arow[c * 32 + 16 + khalf + i] * s);
        }
#pragma unroll
        for (int nt = 0; nt < NT; ++nt) {
            const v16h b = Bp[(c * NT + nt) * 32 + lane];
            acc[nt] = __builtin_amdgcn_wmma_f32_16x16x32_f16(
                /*neg_a=*/false, a, /*neg_b=*/false, b,
                /*c_mod=*/(short)0, acc[nt], /*reuse_a=*/false, /*reuse_b=*/false);
        }
    }

    const int orow0 = (mt << 4) + ((lane >> 4) << 3);
#pragma unroll
    for (int nt = 0; nt < NT; ++nt) {
        const int ocol = nt * 16 + (lane & 15);
        const float bval = bias ? bias[ocol] : 0.0f;
#pragma unroll
        for (int v = 0; v < 8; ++v) {
            float val = acc[nt][v] + bval;
            if (relu) val = fmaxf(val, 0.0f);
            C[(size_t)(orow0 + v) * N + ocol] = val;
        }
    }
}

// ---------------------------------------------------------------------------
// Edge scatter at feature width 64:
//   Out[dst[e]] += gscale[src[e]] * X[src[e]]   (gscale optional)
// 2 floats per thread; rows are 256B-aligned so float2 loads are 8B-aligned.
// All traffic is L2-resident (100k x 64 f32 = 25.6MB << 192MB L2).
// ---------------------------------------------------------------------------
__global__ void edge_scatter64(const float* __restrict__ X, const float* __restrict__ gscale,
                               const int* __restrict__ src, const int* __restrict__ dst,
                               float* __restrict__ Out, long long total_pairs) {
    long long tid = blockIdx.x * (long long)blockDim.x + threadIdx.x;
    if (tid >= total_pairs) return;
    const int e = (int)(tid >> 5);
    const int f = (int)(tid & 31) << 1;
    const int s = src[e];
    const int d = dst[e];
    const float2 v = *(const float2*)(X + (size_t)s * 64 + f);
    const float sc = gscale ? gscale[s] : 1.0f;
    atomicAdd(&Out[(size_t)d * 64 + f],     v.x * sc);
    atomicAdd(&Out[(size_t)d * 64 + f + 1], v.y * sc);
}

// ---------------------------------------------------------------------------
// Pointwise: Out = relu(norm[row] * Min + bias[col]) on [M x 64]
// ---------------------------------------------------------------------------
__global__ void pw_norm_bias_relu(const float* __restrict__ Min, const float* __restrict__ norm,
                                  const float* __restrict__ bias, float* __restrict__ Out,
                                  int M) {
    int tid = blockIdx.x * blockDim.x + threadIdx.x;
    if (tid >= M * 64) return;
    const int row = tid >> 6;
    const int col = tid & 63;
    Out[tid] = fmaxf(fmaf(norm[row], Min[tid], bias[col]), 0.0f);
}

// ---------------------------------------------------------------------------
// Graph readout: hg[graph_ids[row]] += H[row]  over [M x 64]
// ---------------------------------------------------------------------------
__global__ void readout_scatter(const float* __restrict__ H, const int* __restrict__ gid,
                                float* __restrict__ hg, int M) {
    int tid = blockIdx.x * blockDim.x + threadIdx.x;
    if (tid >= M * 64) return;
    const int row = tid >> 6;
    const int f = tid & 63;
    atomicAdd(&hg[(size_t)gid[row] * 64 + f], H[tid]);
}

// out[g][c] = sum_j tanh(hg[g][j]) * Wd[j][c] + bd[c]
__global__ void final_dense(const float* __restrict__ hg, const float* __restrict__ Wd,
                            const float* __restrict__ bd, float* __restrict__ out) {
    const int g = blockIdx.x;
    const int c = threadIdx.x;
    if (c >= NUM_CLASSES) return;
    float s = bd[c];
    for (int j = 0; j < 64; ++j)
        s = fmaf(tanhf(hg[g * 64 + j]), Wd[j * NUM_CLASSES + c], s);
    out[g * NUM_CLASSES + c] = s;
}

// ---------------------------------------------------------------------------
// Launch
// ---------------------------------------------------------------------------
extern "C" void kernel_launch(void* const* d_in, const int* in_sizes, int n_in,
                              void* d_out, int out_size, void* d_ws, size_t ws_size,
                              hipStream_t stream) {
    (void)in_sizes; (void)n_in; (void)out_size; (void)ws_size;

    const float* in_feat = (const float*)d_in[0];
    const int*   src     = (const int*)d_in[1];
    const int*   dst     = (const int*)d_in[2];
    const int*   gid     = (const int*)d_in[3];
    const float* W1 = (const float*)d_in[4];  const float* b1 = (const float*)d_in[5];
    const float* W2 = (const float*)d_in[6];  const float* b2 = (const float*)d_in[7];
    const float* W3 = (const float*)d_in[8];  const float* b3 = (const float*)d_in[9];
    const float* Wd = (const float*)d_in[10]; const float* bd = (const float*)d_in[11];
    float* out = (float*)d_out;

    // Workspace carve-up (all 256B aligned). Total ~104 MB.
    char*  ws  = (char*)d_ws;
    size_t off = 0;
    auto alloc = [&](size_t bytes) -> void* {
        void* p = (void*)(ws + off);
        off = (off + bytes + 255) & ~(size_t)255;
        return p;
    };
    float* bufY     = (float*)alloc((size_t)N_NODES * 64 * 4);   // 25.6 MB
    float* bufM     = (float*)alloc((size_t)N_NODES * 64 * 4);   // 25.6 MB
    float* bufH     = (float*)alloc((size_t)N_NODES * 128 * 4);  // 51.2 MB
    float* deg_out  = (float*)alloc((size_t)N_NODES * 4);
    float* deg_in   = (float*)alloc((size_t)N_NODES * 4);
    float* norm_src = (float*)alloc((size_t)N_NODES * 4);
    float* norm_dst = (float*)alloc((size_t)N_NODES * 4);
    float* hg       = (float*)alloc((size_t)N_GRAPHS * 64 * 4);
    _Float16* W1p   = (_Float16*)alloc((size_t)128 * 64 * 2);    // packed f16 weights
    _Float16* W2p   = (_Float16*)alloc((size_t)64 * 128 * 2);
    _Float16* W3p   = (_Float16*)alloc((size_t)128 * 64 * 2);

    const int T = 256;
    const int nodeBlks   = (N_NODES + T - 1) / T;                    // 391
    const int edgeBlks   = (N_EDGES + T - 1) / T;                    // 12500
    const long long pairTotal = (long long)N_EDGES * 32;             // 102.4M
    const int pairBlks   = (int)((pairTotal + T - 1) / T);           // 400000
    const int pwBlks     = (N_NODES * 64 + T - 1) / T;               // 25000
    const int Mtiles     = N_NODES / 16;                             // 6250
    const int gemmBlks   = (Mtiles * 32 + T - 1) / T;                // 782

    // --- degrees & norms ---
    zero_f32<<<256, T, 0, stream>>>(deg_out, N_NODES);
    zero_f32<<<256, T, 0, stream>>>(deg_in,  N_NODES);
    degree_kernel<<<edgeBlks, T, 0, stream>>>(src, dst, deg_out, deg_in, N_EDGES);
    rsqrt_kernel<<<nodeBlks, T, 0, stream>>>(deg_out, norm_src, N_NODES);
    rsqrt_kernel<<<nodeBlks, T, 0, stream>>>(deg_in,  norm_dst, N_NODES);

    // --- pack weights into WMMA B-operand images (once per launch) ---
    pack_weights<128, 64><<<(128 * 64 + T - 1) / T, T, 0, stream>>>(W1, W1p);
    pack_weights<64, 128><<<(64 * 128 + T - 1) / T, T, 0, stream>>>(W2, W2p);
    pack_weights<128, 64><<<(128 * 64 + T - 1) / T, T, 0, stream>>>(W3, W3p);

    // --- Layer 1: GEMM first (128 -> 64), scatter at width 64 ---
    // Y1 = (norm_src ⊙ X) @ W1
    gemm_wmma<128, 4><<<gemmBlks, T, 0, stream>>>(in_feat, W1p, norm_src, nullptr,
                                                  bufY, Mtiles, 0);
    zero_f32<<<4096, T, 0, stream>>>(bufM, (long long)N_NODES * 64);
    edge_scatter64<<<pairBlks, T, 0, stream>>>(bufY, nullptr, src, dst, bufM, pairTotal);
    // H1 = relu(norm_dst ⊙ M1 + b1)  -> bufY
    pw_norm_bias_relu<<<pwBlks, T, 0, stream>>>(bufM, norm_dst, b1, bufY, N_NODES);

    // --- Layer 2: scatter first at width 64, then GEMM (64 -> 128) ---
    zero_f32<<<4096, T, 0, stream>>>(bufM, (long long)N_NODES * 64);
    edge_scatter64<<<pairBlks, T, 0, stream>>>(bufY, norm_src, src, dst, bufM, pairTotal);
    // H2 = relu((norm_dst ⊙ M2) @ W2 + b2) -> bufH
    gemm_wmma<64, 8><<<gemmBlks, T, 0, stream>>>(bufM, W2p, norm_dst, b2,
                                                 bufH, Mtiles, 1);

    // --- Layer 3: GEMM first (128 -> 64), scatter at width 64 ---
    gemm_wmma<128, 4><<<gemmBlks, T, 0, stream>>>(bufH, W3p, norm_src, nullptr,
                                                  bufY, Mtiles, 0);
    zero_f32<<<4096, T, 0, stream>>>(bufM, (long long)N_NODES * 64);
    edge_scatter64<<<pairBlks, T, 0, stream>>>(bufY, nullptr, src, dst, bufM, pairTotal);
    pw_norm_bias_relu<<<pwBlks, T, 0, stream>>>(bufM, norm_dst, b3, bufY, N_NODES);

    // --- Readout ---
    zero_f32<<<16, T, 0, stream>>>(hg, (long long)N_GRAPHS * 64);
    readout_scatter<<<pwBlks, T, 0, stream>>>(bufY, gid, hg, N_NODES);
    final_dense<<<N_GRAPHS, 32, 0, stream>>>(hg, Wd, bd, out);
}
